// PointNet2SetAbstraction_6322191859820
// MI455X (gfx1250) — matrix operand order repaired and checked by reading, not compiled, not run on hardware
//
#include <hip/hip_runtime.h>
#include <hip/hip_bf16.h>

// ---------------------------------------------------------------------------
// PointNet++ MSG set abstraction for gfx1250 (MI455X).
// Pipeline: FPS -> (per scale) ball query -> gather(bf16) ->
//           3x [ WMMA bf16 GEMM + bias + stats | BN(+ReLU) apply ] ->
//           BN + max-over-K + transposed write.
// GEMM: weights staged in LDS (one copy per block), A-tiles preloaded to
// registers, so the WMMA k-loop issues without global-load waits.
// ---------------------------------------------------------------------------

#define NPTS   4096
#define NB     16
#define NPOINT 512
#define CFEAT  64

typedef __attribute__((ext_vector_type(16))) __bf16        v16bf;
typedef __attribute__((ext_vector_type(8)))  float         v8f;
typedef __attribute__((ext_vector_type(8)))  unsigned int  v8u;

__device__ __forceinline__ unsigned short f2bf(float f) {
    unsigned int u = __float_as_uint(f);
    unsigned int r = u + 0x7FFFu + ((u >> 16) & 1u);   // round to nearest even
    return (unsigned short)(r >> 16);
}
__device__ __forceinline__ float bf2f(unsigned short h) {
    return __uint_as_float(((unsigned int)h) << 16);
}
__device__ __forceinline__ void atomAddF(float* p, float v) {
    unsafeAtomicAdd(p, v);   // native global_atomic_add_f32
}

// ---------------------------------------------------------------------------
// 1) Farthest point sampling: one block per batch, xyz staged in LDS.
//    Writes new_xyz (B, NPOINT, 3) to the first region of d_out.
// ---------------------------------------------------------------------------
__global__ __launch_bounds__(256) void fps_kernel(const float* __restrict__ xyz,
                                                  float* __restrict__ out_xyz) {
    __shared__ float px[NPTS], py[NPTS], pz[NPTS];
    __shared__ float rval[256];
    __shared__ int   ridx[256];
    __shared__ float curx, cury, curz;

    const int b = blockIdx.x;
    const int t = threadIdx.x;
    const float* base = xyz + (size_t)b * NPTS * 3;
    for (int i = t; i < NPTS; i += 256) {
        px[i] = base[3 * i + 0];
        py[i] = base[3 * i + 1];
        pz[i] = base[3 * i + 2];
    }
    float mind[16];
#pragma unroll
    for (int j = 0; j < 16; ++j) mind[j] = 1e10f;
    __syncthreads();

    if (t == 0) {
        curx = px[0]; cury = py[0]; curz = pz[0];
        float* o = out_xyz + (size_t)b * NPOINT * 3;
        o[0] = px[0]; o[1] = py[0]; o[2] = pz[0];
    }
    __syncthreads();

    for (int it = 1; it < NPOINT; ++it) {
        const float cx = curx, cy = cury, cz = curz;
        float best = -1.0f;
        int   bi   = 0;
#pragma unroll
        for (int j = 0; j < 16; ++j) {
            const int p = t + 256 * j;
            const float dx = px[p] - cx, dy = py[p] - cy, dz = pz[p] - cz;
            const float d = dx * dx + dy * dy + dz * dz;
            mind[j] = fminf(mind[j], d);
            if (mind[j] > best) { best = mind[j]; bi = p; }
        }
        rval[t] = best; ridx[t] = bi;
        __syncthreads();
        for (int s = 128; s > 0; s >>= 1) {
            if (t < s) {
                if (rval[t + s] > rval[t] ||
                    (rval[t + s] == rval[t] && ridx[t + s] < ridx[t])) {
                    rval[t] = rval[t + s]; ridx[t] = ridx[t + s];
                }
            }
            __syncthreads();
        }
        if (t == 0) {
            const int c = ridx[0];
            curx = px[c]; cury = py[c]; curz = pz[c];
            float* o = out_xyz + ((size_t)b * NPOINT + it) * 3;
            o[0] = curx; o[1] = cury; o[2] = curz;
        }
        __syncthreads();
    }
}

// ---------------------------------------------------------------------------
// 2) Ball query (first NS neighbors by ascending index within radius).
// ---------------------------------------------------------------------------
template <int NS>
__global__ __launch_bounds__(256) void ballq_kernel(const float* __restrict__ xyz,
                                                    const float* __restrict__ newxyz,
                                                    int* __restrict__ idxOut,
                                                    float r2) {
    __shared__ float px[NPTS], py[NPTS], pz[NPTS];
    const int b = blockIdx.y;
    const int s = blockIdx.x * 256 + threadIdx.x;
    const float* base = xyz + (size_t)b * NPTS * 3;
    for (int i = threadIdx.x; i < NPTS; i += 256) {
        px[i] = base[3 * i + 0];
        py[i] = base[3 * i + 1];
        pz[i] = base[3 * i + 2];
    }
    __syncthreads();

    const float* q = newxyz + ((size_t)b * NPOINT + s) * 3;
    const float qx = q[0], qy = q[1], qz = q[2];
    int* out = idxOut + ((size_t)b * NPOINT + s) * NS;
    int cnt = 0, first = NPTS - 1;
    for (int j = 0; j < NPTS && cnt < NS; ++j) {
        const float dx = px[j] - qx, dy = py[j] - qy, dz = pz[j] - qz;
        if (dx * dx + dy * dy + dz * dz < r2) {
            if (cnt == 0) first = j;
            out[cnt++] = j;
        }
    }
    for (int k = cnt; k < NS; ++k) out[k] = first;
}

// ---------------------------------------------------------------------------
// 3) Gather grouped input x0 = [rel_xyz(3) | feat(64) | pad->96] as bf16.
// ---------------------------------------------------------------------------
__global__ __launch_bounds__(256) void gather_kernel(const float* __restrict__ xyz,
                                                     const float* __restrict__ newxyz,
                                                     const float* __restrict__ feat,
                                                     const int* __restrict__ idx,
                                                     unsigned short* __restrict__ X0,
                                                     int K) {
    const size_t r = (size_t)blockIdx.x * 256 + threadIdx.x;   // over B*NPOINT*K
    const int b = (int)(r / ((size_t)NPOINT * K));
    const int s = (int)((r / K) % NPOINT);
    const int i = idx[r];
    const float* p = xyz + ((size_t)b * NPTS + i) * 3;
    const float* q = newxyz + ((size_t)b * NPOINT + s) * 3;
    unsigned short* x = X0 + r * 96;
    x[0] = f2bf(p[0] - q[0]);
    x[1] = f2bf(p[1] - q[1]);
    x[2] = f2bf(p[2] - q[2]);
    const float* fb = feat + (size_t)b * CFEAT * NPTS + i;
#pragma unroll
    for (int c = 0; c < CFEAT; ++c) x[3 + c] = f2bf(fb[(size_t)c * NPTS]);
#pragma unroll
    for (int c = 3 + CFEAT; c < 96; ++c) x[c] = 0;
}

// ---------------------------------------------------------------------------
// 4) Weight prep: f32 (O,C) -> bf16 padded (O,CP); also zero the stats buffer.
// ---------------------------------------------------------------------------
__global__ __launch_bounds__(256) void prepw_kernel(const float* __restrict__ W,
                                                    unsigned short* __restrict__ Wp,
                                                    float* __restrict__ sums,
                                                    int O, int C, int CP) {
    const int t = blockIdx.x * 256 + threadIdx.x;
    if (t < O * CP) {
        const int o = t / CP, c = t % CP;
        Wp[t] = (c < C) ? f2bf(W[o * C + c]) : (unsigned short)0;
    }
    if (t < 2 * O) sums[t] = 0.0f;
}

// ---------------------------------------------------------------------------
// 5) GEMM (bf16 WMMA, f32 accum) + bias + bf16 store + per-channel sum/sumsq.
//    X: (R, CIN) bf16, Wp: (COUT, CIN) bf16, Y: (R, COUT) bf16.
//    Block = 256 threads = 8 waves; each wave owns a 16-row stripe x full COUT.
//    Weights staged once per block in LDS; A k-tiles preloaded to registers.
// ---------------------------------------------------------------------------
template <int CIN, int COUT>
__global__ __launch_bounds__(256) void gemm_kernel(const unsigned short* __restrict__ X,
                                                   const unsigned short* __restrict__ Wp,
                                                   const float* __restrict__ bias,
                                                   unsigned short* __restrict__ Y,
                                                   float* __restrict__ sums,
                                                   int R) {
    constexpr int NT = COUT / 16;   // N tiles
    constexpr int KT = CIN / 32;    // K steps
    __shared__ unsigned short wlds[COUT * CIN];   // <= 24 KB
    __shared__ float lsum[2 * COUT];

    const int tid  = threadIdx.x;
    const int wave = tid >> 5;
    const int lane = tid & 31;
    const int half = lane >> 4;     // 0: lanes 0-15, 1: lanes 16-31
    const int ln   = lane & 15;

    // Cooperative stage of the whole weight matrix into LDS (b128 copies).
    {
        const uint4* src = (const uint4*)Wp;
        uint4*       dst = (uint4*)wlds;
#pragma unroll
        for (int i = tid; i < (COUT * CIN) / 8; i += 256) dst[i] = src[i];
    }
    for (int i = tid; i < 2 * COUT; i += 256) lsum[i] = 0.0f;
    __syncthreads();

    const int rowBase = blockIdx.x * 128 + wave * 16;
    const int rowA    = rowBase + ln;           // A-matrix row held by this lane

    // Preload all KT A-tiles for this wave's 16-row stripe.
    // A tile (16x32 bf16): VGPR g<4 -> K = 2g + 8*half, g>=4 -> K = 16+2*(g-4)+8*half
    v8u au[KT];
#pragma unroll
    for (int kt = 0; kt < KT; ++kt) {
        const unsigned short* ap = X + (size_t)rowA * CIN + kt * 32 + 8 * half;
#pragma unroll
        for (int g = 0; g < 4; ++g)
            au[kt][g] = *(const unsigned int*)(ap + 2 * g);
#pragma unroll
        for (int g = 0; g < 4; ++g)
            au[kt][4 + g] = *(const unsigned int*)(ap + 16 + 2 * g);
    }

    v8f acc[NT];
#pragma unroll
    for (int nt = 0; nt < NT; ++nt) {
#pragma unroll
        for (int i = 0; i < 8; ++i) acc[nt][i] = 0.0f;
    }

#pragma unroll
    for (int kt = 0; kt < KT; ++kt) {
        const v16bf a = __builtin_bit_cast(v16bf, au[kt]);
#pragma unroll
        for (int nt = 0; nt < NT; ++nt) {
            // B tile (32x16 bf16) from LDS: VGPR g<4 -> K = 2g + 8*half, g>=4 -> K = 16+2*(g-4)+8*half
            const int o = nt * 16 + ln;
            const unsigned short* bp = wlds + o * CIN + kt * 32 + 8 * half;
            v8u bu;
#pragma unroll
            for (int g = 0; g < 8; ++g)
                bu[g] = *(const unsigned int*)(bp + 2 * g + ((g >> 2) << 3));
            const v16bf bmat = __builtin_bit_cast(v16bf, bu);
            acc[nt] = __builtin_amdgcn_wmma_f32_16x16x32_bf16(
                false, a, false, bmat, (short)0, acc[nt], false, false);
        }
    }

    // Epilogue: bias, bf16 store, per-channel stats.
    // C/D layout: element v of lane -> row = rowBase + v + 8*half, chan = nt*16 + ln.
#pragma unroll
    for (int nt = 0; nt < NT; ++nt) {
        const int o = nt * 16 + ln;
        const float bv = bias[o];
        float s1 = 0.0f, s2 = 0.0f;
#pragma unroll
        for (int v = 0; v < 8; ++v) {
            const float y = acc[nt][v] + bv;
            const int   r = rowBase + v + 8 * half;
            Y[(size_t)r * COUT + o] = f2bf(y);
            s1 += y;
            s2 += y * y;
        }
        s1 += __shfl_xor(s1, 16, 32);
        s2 += __shfl_xor(s2, 16, 32);
        if (half == 0) {
            atomicAdd(&lsum[o], s1);          // ds_add_f32
            atomicAdd(&lsum[COUT + o], s2);
        }
    }
    __syncthreads();
    for (int i = tid; i < 2 * COUT; i += 256) atomAddF(&sums[i], lsum[i]);
}

// ---------------------------------------------------------------------------
// 6) BatchNorm (+ReLU) apply: bf16 in, bf16 out (input of next layer).
// ---------------------------------------------------------------------------
__global__ __launch_bounds__(256) void bn_apply_kernel(const unsigned short* __restrict__ Y,
                                                       const float* __restrict__ sums,
                                                       const float* __restrict__ g,
                                                       const float* __restrict__ beta,
                                                       unsigned short* __restrict__ Xn,
                                                       int R, int COUT) {
    const size_t i = (size_t)blockIdx.x * 256 + threadIdx.x;
    if (i >= (size_t)R * COUT) return;
    const int o = (int)(i % COUT);
    const float cnt = (float)R;
    const float mu  = sums[o] / cnt;
    const float var = sums[COUT + o] / cnt - mu * mu;
    const float sc  = g[o] * rsqrtf(var + 1e-5f);
    float y = (bf2f(Y[i]) - mu) * sc + beta[o];
    y = fmaxf(y, 0.0f);
    Xn[i] = f2bf(y);
}

// ---------------------------------------------------------------------------
// 7) Final layer: BatchNorm, max over K, write (B, 256, NPOINT) transposed.
// ---------------------------------------------------------------------------
__global__ __launch_bounds__(256) void bn_max_kernel(const unsigned short* __restrict__ Y,
                                                     const float* __restrict__ sums,
                                                     const float* __restrict__ g,
                                                     const float* __restrict__ beta,
                                                     float* __restrict__ out_feat,
                                                     int K, int COUT, int chanBase, int R) {
    const int t = blockIdx.x * 256 + threadIdx.x;   // over B*NPOINT*COUT
    if (t >= NB * NPOINT * COUT) return;
    const int o = t % COUT;
    const int s = (t / COUT) % NPOINT;
    const int b = t / (COUT * NPOINT);

    const float cnt = (float)R;
    const float mu  = sums[o] / cnt;
    const float var = sums[COUT + o] / cnt - mu * mu;
    const float sc  = g[o] * rsqrtf(var + 1e-5f);
    const float sh  = beta[o] - mu * sc;

    float m = -3.4e38f;
    const size_t rowBase = ((size_t)b * NPOINT + s) * K;
    for (int k = 0; k < K; ++k) {
        const float y = bf2f(Y[(rowBase + k) * COUT + o]) * sc + sh;
        m = fmaxf(m, y);
    }
    out_feat[((size_t)b * 256 + chanBase + o) * NPOINT + s] = m;
}

// ---------------------------------------------------------------------------
// Host-side dispatch
// ---------------------------------------------------------------------------
static void launch_gemm(int CP, int O,
                        const unsigned short* X, const unsigned short* Wp,
                        const float* bias, unsigned short* Y, float* sums,
                        int R, hipStream_t st) {
    const dim3 grid((unsigned)(R / 128));
    if (CP == 96 && O == 64)
        gemm_kernel<96, 64><<<grid, 256, 0, st>>>(X, Wp, bias, Y, sums, R);
    else if (CP == 64 && O == 64)
        gemm_kernel<64, 64><<<grid, 256, 0, st>>>(X, Wp, bias, Y, sums, R);
    else if (CP == 64 && O == 128)
        gemm_kernel<64, 128><<<grid, 256, 0, st>>>(X, Wp, bias, Y, sums, R);
    else if (CP == 64 && O == 96)
        gemm_kernel<64, 96><<<grid, 256, 0, st>>>(X, Wp, bias, Y, sums, R);
    else if (CP == 96 && O == 128)
        gemm_kernel<96, 128><<<grid, 256, 0, st>>>(X, Wp, bias, Y, sums, R);
}

extern "C" void kernel_launch(void* const* d_in, const int* in_sizes, int n_in,
                              void* d_out, int out_size, void* d_ws, size_t ws_size,
                              hipStream_t stream) {
    (void)in_sizes; (void)n_in; (void)out_size; (void)ws_size;

    const float* xyz  = (const float*)d_in[0];
    const float* feat = (const float*)d_in[1];
    const float* Wm[2][3]; const float* Bv[2][3];
    const float* Gv[2][3]; const float* Bt[2][3];
    int pi = 2;
    for (int s = 0; s < 2; ++s)
        for (int l = 0; l < 3; ++l) {
            Wm[s][l] = (const float*)d_in[pi++];
            Bv[s][l] = (const float*)d_in[pi++];
            Gv[s][l] = (const float*)d_in[pi++];
            Bt[s][l] = (const float*)d_in[pi++];
        }

    char* ws = (char*)d_ws;
    const size_t maxR = (size_t)NB * NPOINT * 64;                 // 524288
    unsigned short* bufA = (unsigned short*)(ws);                 // maxR*128 bf16
    unsigned short* bufB = (unsigned short*)(ws + maxR * 128 * 2);
    int*   idxBuf = (int*)  (ws + 2 * maxR * 128 * 2);
    unsigned short* Wp = (unsigned short*)(ws + 2 * maxR * 128 * 2 + maxR * 4);
    float* sums   = (float*)(ws + 2 * maxR * 128 * 2 + maxR * 4 + 32768);

    float* out_xyz  = (float*)d_out;                              // (B, 512, 3)
    float* out_feat = (float*)d_out + (size_t)NB * NPOINT * 3;    // (B, 256, 512)

    // 1) FPS -> new_xyz
    fps_kernel<<<16, 256, 0, stream>>>(xyz, out_xyz);

    const int   Ks[2]   = {32, 64};
    const float rad[2]  = {0.2f, 0.4f};
    const int   dims[2][4] = {{67, 64, 64, 128}, {67, 64, 96, 128}};
    const int   cpad[2][3] = {{96, 64, 64}, {96, 64, 96}};

    for (int s = 0; s < 2; ++s) {
        const int K = Ks[s];
        const int R = NB * NPOINT * K;
        const float r2 = rad[s] * rad[s];

        if (K == 32)
            ballq_kernel<32><<<dim3(2, 16), 256, 0, stream>>>(xyz, out_xyz, idxBuf, r2);
        else
            ballq_kernel<64><<<dim3(2, 16), 256, 0, stream>>>(xyz, out_xyz, idxBuf, r2);

        gather_kernel<<<(unsigned)(R / 256), 256, 0, stream>>>(xyz, out_xyz, feat,
                                                               idxBuf, bufA, K);

        for (int l = 0; l < 3; ++l) {
            const int C  = dims[s][l];
            const int CP = cpad[s][l];
            const int O  = dims[s][l + 1];
            const int pt = O * CP > 2 * O ? O * CP : 2 * O;
            prepw_kernel<<<(unsigned)((pt + 255) / 256), 256, 0, stream>>>(
                Wm[s][l], Wp, sums, O, C, CP);
            launch_gemm(CP, O, bufA, Wp, Bv[s][l], bufB, sums, R, stream);
            if (l < 2) {
                bn_apply_kernel<<<(unsigned)(((size_t)R * O + 255) / 256), 256, 0, stream>>>(
                    bufB, sums, Gv[s][l], Bt[s][l], bufA, R, O);
            } else {
                bn_max_kernel<<<(unsigned)((NB * NPOINT * O + 255) / 256), 256, 0, stream>>>(
                    bufB, sums, Gv[s][l], Bt[s][l], out_feat, K, O, s * 128, R);
            }
        }
    }
}
